// BatchPrefillWrapper_77945066488431
// MI455X (gfx1250) — compile-verified, matrix operand order checked by reading, not run on hardware
//
#include <hip/hip_runtime.h>

// ---------------------------------------------------------------------------
// MI455X (gfx1250) implementation of the hybrid (delta-rule + attention) block.
// Big GEMMs + attention: bf16 WMMA (v_wmma_f32_16x16x32_bf16), f32 accumulate.
// Delta-rule scan: f32 VALU, LDS-resident state, async global->LDS staging.
// ---------------------------------------------------------------------------

typedef __bf16 bh;
typedef __attribute__((ext_vector_type(8)))  __bf16 bh8;
typedef __attribute__((ext_vector_type(16))) __bf16 bh16;
typedef __attribute__((ext_vector_type(8)))  float  f8;
typedef __attribute__((ext_vector_type(4)))  int    v4i;

#define S_LEN 1024
#define HID   2048
#define CDIM  4096
#define DFF   6144
#define NVH   16      // v heads
#define NKH   8       // qk heads
#define HD    128
#define MAXKV 4096
#define NQH   16
#define NKVH  2

#if defined(__AMDGCN__) && __has_builtin(__builtin_amdgcn_global_load_async_to_lds_b128) && __has_builtin(__builtin_amdgcn_s_wait_asynccnt)
#define USE_ASYNC_LDS 1
#endif

__device__ __forceinline__ void cp_f4(const float* g, float* l) {
#ifdef USE_ASYNC_LDS
  __builtin_amdgcn_global_load_async_to_lds_b128(
      (__attribute__((address_space(1))) v4i*)g,
      (__attribute__((address_space(3))) v4i*)l, 0, 0);
#else
  *(float4*)l = *(const float4*)g;
#endif
}
__device__ __forceinline__ void cp_wait() {
#ifdef USE_ASYNC_LDS
  __builtin_amdgcn_s_wait_asynccnt(0);
#endif
}

__device__ __forceinline__ bh16 mk16(bh8 lo, bh8 hi) {
  bh16 r;
#pragma unroll
  for (int i = 0; i < 8; ++i) { r[i] = lo[i]; r[i + 8] = hi[i]; }
  return r;
}

__device__ __forceinline__ float sigf(float x) { return 1.f / (1.f + __expf(-x)); }

// ---------------------------------------------------------------------------
// Generic WMMA GEMM: C[M,N] = A[M,K] * B[K,N], f32 in/out, bf16 compute.
// MODE: 0 store, 1 C += acc (residual), 2 silu(acc), 3 acc * aux[idx]
// Tile 128x128x32, 256 threads (8 wave32), wave grid 2(M) x 4(N).
// ---------------------------------------------------------------------------
template <int MODE>
__global__ __launch_bounds__(256)
void gemm_k(const float* __restrict__ A, const float* __restrict__ B,
            float* __restrict__ C, const float* __restrict__ aux,
            int M, int N, int K) {
  __shared__ __align__(16) bh As[128][40];  // stride 40 -> conflict-free b128 rows
  __shared__ __align__(16) bh Bs[128][40];  // stored transposed: Bs[n][k]
  const int t = threadIdx.x;
  const int lane = t & 31, wv = t >> 5;
  const int wm = wv >> 2, wn = wv & 3;
  const int l16 = lane & 15, hf = lane >> 4;
  const int m0 = blockIdx.y * 128, n0 = blockIdx.x * 128;

  f8 acc[4][2];
#pragma unroll
  for (int a = 0; a < 4; ++a)
#pragma unroll
    for (int b = 0; b < 2; ++b)
#pragma unroll
      for (int e = 0; e < 8; ++e) acc[a][b][e] = 0.f;

  const int arow = t >> 3, aq = t & 7;   // A: 32 rows/pass, 8 float4 per row
  const int bkr = t >> 5, bnq = t & 31;  // B: 8 k-rows/pass, 32 float4 across N

  for (int k0 = 0; k0 < K; k0 += 32) {
    if (k0 + 32 < K) {  // global_prefetch_b8 next tiles
      __builtin_prefetch(A + (size_t)(m0 + arow) * K + k0 + 32, 0, 3);
      __builtin_prefetch(B + (size_t)(k0 + 32 + bkr) * N + n0 + bnq * 4, 0, 3);
    }
#pragma unroll
    for (int p = 0; p < 4; ++p) {
      const float4 v = *(const float4*)(A + (size_t)(m0 + arow + p * 32) * K + k0 + aq * 4);
      bh* d = &As[arow + p * 32][aq * 4];
      d[0] = (bh)v.x; d[1] = (bh)v.y; d[2] = (bh)v.z; d[3] = (bh)v.w;
    }
#pragma unroll
    for (int p = 0; p < 4; ++p) {
      const float4 v = *(const float4*)(B + (size_t)(k0 + bkr + p * 8) * N + n0 + bnq * 4);
      Bs[bnq * 4 + 0][bkr + p * 8] = (bh)v.x;
      Bs[bnq * 4 + 1][bkr + p * 8] = (bh)v.y;
      Bs[bnq * 4 + 2][bkr + p * 8] = (bh)v.z;
      Bs[bnq * 4 + 3][bkr + p * 8] = (bh)v.w;
    }
    __syncthreads();
    bh16 af[4], bfrag[2];
#pragma unroll
    for (int mf = 0; mf < 4; ++mf) {
      const int r = wm * 64 + mf * 16 + l16;
      af[mf] = mk16(*(const bh8*)&As[r][hf * 8], *(const bh8*)&As[r][16 + hf * 8]);
    }
#pragma unroll
    for (int nf = 0; nf < 2; ++nf) {
      const int cidx = wn * 32 + nf * 16 + l16;
      bfrag[nf] = mk16(*(const bh8*)&Bs[cidx][hf * 8], *(const bh8*)&Bs[cidx][16 + hf * 8]);
    }
#pragma unroll
    for (int mf = 0; mf < 4; ++mf)
#pragma unroll
      for (int nf = 0; nf < 2; ++nf)
        acc[mf][nf] = __builtin_amdgcn_wmma_f32_16x16x32_bf16(
            false, af[mf], false, bfrag[nf], (short)0, acc[mf][nf], false, false);
    __syncthreads();
  }
#pragma unroll
  for (int mf = 0; mf < 4; ++mf)
#pragma unroll
    for (int nf = 0; nf < 2; ++nf) {
      const int col = n0 + wn * 32 + nf * 16 + l16;
#pragma unroll
      for (int r = 0; r < 8; ++r) {
        const int row = m0 + wm * 64 + mf * 16 + r + 8 * hf;
        const size_t idx = (size_t)row * N + col;
        float v = acc[mf][nf][r];
        if (MODE == 0) C[idx] = v;
        else if (MODE == 1) C[idx] += v;
        else if (MODE == 2) C[idx] = v * sigf(v);
        else C[idx] = v * aux[idx];
      }
    }
}

// ---------------------------------------------------------------------------
// Elementwise / small kernels
// ---------------------------------------------------------------------------
__global__ __launch_bounds__(256)
void embed_k(const int* __restrict__ ids, const float* __restrict__ ew,
             float* __restrict__ hid) {
  const int s = blockIdx.x;
  const int id = ids[s];
  for (int d = threadIdx.x; d < HID; d += 256)
    hid[(size_t)s * HID + d] = ew[(size_t)id * HID + d];
}

__global__ __launch_bounds__(256)
void rmsnorm_k(const float* __restrict__ x, const float* __restrict__ w,
               float* __restrict__ y, int dim) {
  const int s = blockIdx.x;
  __shared__ float red[8];
  float sq = 0.f;
  for (int d = threadIdx.x; d < dim; d += 256) {
    float v = x[(size_t)s * dim + d];
    sq += v * v;
  }
  for (int m = 1; m < 32; m <<= 1) sq += __shfl_xor(sq, m);
  if ((threadIdx.x & 31) == 0) red[threadIdx.x >> 5] = sq;
  __syncthreads();
  float tot = 0.f;
#pragma unroll
  for (int i = 0; i < 8; ++i) tot += red[i];
  const float inv = rsqrtf(tot / dim + 1e-6f);
  for (int d = threadIdx.x; d < dim; d += 256)
    y[(size_t)s * dim + d] = x[(size_t)s * dim + d] * inv * w[d];
}

// depthwise causal conv (K=4) + silu.  padded = [conv_state(4) | mixed_pre^T]
__global__ __launch_bounds__(256)
void conv_k(const float* __restrict__ mpre, const float* __restrict__ cst,
            const float* __restrict__ cw, const float* __restrict__ cb,
            float* __restrict__ mixed) {
  const int d = blockIdx.x * 256 + threadIdx.x;  // 4096
  const int s = blockIdx.y;                      // 1024
  float acc = cb[d];
#pragma unroll
  for (int k = 0; k < 4; ++k) {
    const int tt = s + 1 + k;
    const float pv = (tt < 4) ? cst[d * 4 + tt]
                              : mpre[(size_t)(tt - 4) * CDIM + d];
    acc += cw[d * 4 + k] * pv;
  }
  mixed[(size_t)s * CDIM + d] = acc * sigf(acc);
}

__global__ __launch_bounds__(256)
void newconv_k(const float* __restrict__ mpre, float* __restrict__ ncv) {
  const int d = blockIdx.x * 256 + threadIdx.x;
#pragma unroll
  for (int k = 0; k < 4; ++k)
    ncv[d * 4 + k] = mpre[(size_t)(S_LEN - 4 + k) * CDIM + d];
}

// beta = sigmoid(x @ bw); g = -exp(A_log)*softplus(x @ aw + dt_bias)   (N=16)
__global__ __launch_bounds__(256)
void ba_k(const float* __restrict__ x, const float* __restrict__ bw,
          const float* __restrict__ aw, const float* __restrict__ alog,
          const float* __restrict__ dtb, float* __restrict__ beta,
          float* __restrict__ g) {
  const int s = blockIdx.x;
  const int n = threadIdx.x & 15, part = threadIdx.x >> 4;
  __shared__ float rb[16][16], ra[16][16];
  float sb = 0.f, sa = 0.f;
  for (int k = part * 128; k < part * 128 + 128; ++k) {
    const float xv = x[(size_t)s * HID + k];
    sb += xv * bw[k * 16 + n];
    sa += xv * aw[k * 16 + n];
  }
  rb[part][n] = sb; ra[part][n] = sa;
  __syncthreads();
  if (threadIdx.x < 16) {
    const int nn = threadIdx.x;
    float b = 0.f, a = 0.f;
#pragma unroll
    for (int p = 0; p < 16; ++p) { b += rb[p][nn]; a += ra[p][nn]; }
    beta[s * 16 + nn] = sigf(b);
    const float ad = a + dtb[nn];
    const float sp = (ad > 20.f) ? ad : log1pf(__expf(ad));
    g[s * 16 + nn] = -__expf(alog[nn]) * sp;
  }
}

// ---------------------------------------------------------------------------
// Gated delta rule. grid = 64 blocks: (v-head h, Dv quarter dh). 256 threads.
// Sequential over 16 chunks of 64; state (128 x 32) lives in LDS.
// q/k/v chunk tiles staged with async global->LDS loads (ASYNCcnt path).
// ---------------------------------------------------------------------------
#define DVH 32
__global__ __launch_bounds__(256)
void delta_k(const float* __restrict__ mixed, const float* __restrict__ gtok,
             const float* __restrict__ btok, const float* __restrict__ st0,
             float* __restrict__ core, float* __restrict__ recout) {
  const int h = blockIdx.x >> 2;
  const int dh = blockIdx.x & 3;
  const int hq = h >> 1;  // jnp.repeat(rep=2)
  const int t = threadIdx.x;

  __shared__ __align__(16) float st[128][DVH + 4];
  __shared__ __align__(16) float qn[64][132];
  __shared__ __align__(16) float kn[64][132];
  __shared__ __align__(16) float vv[64][DVH + 4];
  __shared__ float Am[64][68];
  __shared__ float wb[64][DVH + 4];
  __shared__ __align__(16) float kcg[64][132];
  __shared__ float gch[64], bch[64], bg[64], egc[64], ef[64];

  for (int i = t; i < 128 * DVH; i += 256) {
    const int k = i / DVH, d = i % DVH;
    st[k][d] = st0[((size_t)h * 128 + k) * 128 + dh * DVH + d];
  }
  __syncthreads();

  for (int c = 0; c < 16; ++c) {
    const int s0 = c * 64;
    for (int i = t; i < 64 * 32; i += 256) {
      const int r = i >> 5, q = i & 31;
      cp_f4(mixed + (size_t)(s0 + r) * CDIM + hq * HD + q * 4, &qn[r][q * 4]);
      cp_f4(mixed + (size_t)(s0 + r) * CDIM + 1024 + hq * HD + q * 4, &kn[r][q * 4]);
    }
    for (int i = t; i < 64 * (DVH / 4); i += 256) {
      const int r = i / (DVH / 4), q = i % (DVH / 4);
      cp_f4(mixed + (size_t)(s0 + r) * CDIM + 2048 + h * HD + dh * DVH + q * 4,
            &vv[r][q * 4]);
    }
    cp_wait();
    __syncthreads();

    {  // L2 normalize q (and scale Dk^-0.5) and k; 4 threads per row
      const int r = t >> 2, pp = t & 3;
      float sq = 0.f, sk = 0.f;
      for (int d = pp * 32; d < pp * 32 + 32; ++d) {
        float a = qn[r][d]; sq += a * a;
        float b = kn[r][d]; sk += b * b;
      }
      sq += __shfl_xor(sq, 1); sq += __shfl_xor(sq, 2);
      sk += __shfl_xor(sk, 1); sk += __shfl_xor(sk, 2);
      const float qi = rsqrtf(sq + 1e-6f) * 0.08838834764831845f;
      const float ki = rsqrtf(sk + 1e-6f);
      for (int d = pp * 32; d < pp * 32 + 32; ++d) { qn[r][d] *= qi; kn[r][d] *= ki; }
    }
    if (t == 0) {
      float a = 0.f;
      for (int i = 0; i < 64; ++i) { a += gtok[(size_t)(s0 + i) * 16 + h]; gch[i] = a; }
    }
    if (t >= 64 && t < 128) bch[t - 64] = btok[(size_t)(s0 + (t - 64)) * 16 + h];
    __syncthreads();

    if (t < 64) { egc[t] = __expf(gch[t]); bg[t] = bch[t] * egc[t]; }
    // A[i][j] = -(beta_i k_i.k_j) exp(g_i-g_j), strictly lower
    for (int idx = t; idx < 64 * 64; idx += 256) {
      const int i = idx >> 6, j = idx & 63;
      float v = 0.f;
      if (i > j) {
        float dt_ = 0.f;
        for (int d = 0; d < 128; ++d) dt_ += kn[i][d] * kn[j][d];
        v = -(bch[i] * dt_) * __expf(gch[i] - gch[j]);
      }
      Am[i][j] = v;
    }
    __syncthreads();

    // (I - A) w = v*beta ; (I - A) kc = k*beta*exp(g)  (forward substitution,
    // columns independent -> one thread per column, no barriers)
    if (t < DVH) {
      const int d = t;
      for (int i = 0; i < 64; ++i) {
        float s = vv[i][d] * bch[i];
        for (int j = 0; j < i; ++j) s += Am[i][j] * wb[j][d];
        wb[i][d] = s;
      }
    } else if (t < DVH + 128) {
      const int d = t - DVH;
      for (int i = 0; i < 64; ++i) {
        float s = kn[i][d] * bg[i];
        for (int j = 0; j < i; ++j) s += Am[i][j] * kcg[j][d];
        kcg[i][d] = s;
      }
    }
    __syncthreads();

    // attn (reuse Am) + decay factors for state update
    if (t < 64) ef[t] = __expf(gch[63] - gch[t]);
    for (int idx = t; idx < 64 * 64; idx += 256) {
      const int i = idx >> 6, j = idx & 63;
      float v = 0.f;
      if (i >= j) {
        float dt_ = 0.f;
        for (int d = 0; d < 128; ++d) dt_ += qn[i][d] * kn[j][d];
        v = dt_ * __expf(gch[i] - gch[j]);
      }
      Am[i][j] = v;
    }
    __syncthreads();

    // v_new = w - kc @ state (in place) ; scale kn by exp(g_last - g_i)
    for (int idx = t; idx < 64 * DVH; idx += 256) {
      const int i = idx / DVH, d = idx % DVH;
      float s = 0.f;
      for (int k = 0; k < 128; ++k) s += kcg[i][k] * st[k][d];
      wb[i][d] -= s;
    }
    for (int idx = t; idx < 64 * 128; idx += 256) {
      const int i = idx >> 7, d = idx & 127;
      kn[i][d] *= ef[i];
    }
    __syncthreads();

    // out = (q e^g) @ state + attn @ v_new
    for (int idx = t; idx < 64 * DVH; idx += 256) {
      const int i = idx / DVH, d = idx % DVH;
      float s = 0.f;
      for (int k = 0; k < 128; ++k) s += qn[i][k] * st[k][d];
      s *= egc[i];
      for (int j = 0; j <= i; ++j) s += Am[i][j] * wb[j][d];
      core[(size_t)(s0 + i) * HID + h * HD + dh * DVH + d] = s;
    }
    __syncthreads();

    // state = state*e^{g_last} + (k e^{g_last-g})^T @ v_new
    const float egl = __expf(gch[63]);
    for (int idx = t; idx < 128 * DVH; idx += 256) {
      const int k = idx / DVH, d = idx % DVH;
      float s = st[k][d] * egl;
      for (int i = 0; i < 64; ++i) s += kn[i][k] * wb[i][d];
      st[k][d] = s;
    }
    __syncthreads();
  }
  for (int i = t; i < 128 * DVH; i += 256) {
    const int k = i / DVH, d = i % DVH;
    recout[((size_t)h * 128 + k) * 128 + dh * DVH + d] = st[k][d];
  }
}

// gated rmsnorm per (s, head): y = (x*silu(z)) * rsqrt(mean(.)+eps) * w
__global__ __launch_bounds__(128)
void gatednorm_k(const float* __restrict__ core, const float* __restrict__ z,
                 const float* __restrict__ w, float* __restrict__ outp) {
  const int h = blockIdx.x, s = blockIdx.y, d = threadIdx.x;
  __shared__ float red[4];
  const size_t idx = (size_t)s * HID + h * HD + d;
  const float zz = z[idx];
  const float y = core[idx] * (zz * sigf(zz));
  float sq = y * y;
  for (int m = 1; m < 32; m <<= 1) sq += __shfl_xor(sq, m);
  if ((threadIdx.x & 31) == 0) red[threadIdx.x >> 5] = sq;
  __syncthreads();
  const float tot = red[0] + red[1] + red[2] + red[3];
  outp[idx] = y * rsqrtf(tot / 128.f + 1e-6f) * w[d];
}

// q head prep: rmsnorm(128) + RoPE -> Qb[h][s][d]
__global__ __launch_bounds__(128)
void qprep_k(const float* __restrict__ qg, const float* __restrict__ w,
             const float* __restrict__ cb, const float* __restrict__ sb,
             float* __restrict__ Qb) {
  const int n = blockIdx.x, s = blockIdx.y, d = threadIdx.x;
  __shared__ float red[4];
  __shared__ float xr[128];
  const float x = qg[(size_t)s * 4096 + n * 256 + d];
  float sq = x * x;
  for (int m = 1; m < 32; m <<= 1) sq += __shfl_xor(sq, m);
  if ((threadIdx.x & 31) == 0) red[threadIdx.x >> 5] = sq;
  __syncthreads();
  const float inv = rsqrtf((red[0] + red[1] + red[2] + red[3]) / 128.f + 1e-6f);
  const float xn = x * inv * w[d];
  xr[d] = xn;
  __syncthreads();
  const float rot = (d < 64) ? -xr[d + 64] : xr[d - 64];
  Qb[((size_t)n * S_LEN + s) * HD + d] = xn * cb[s * HD + d] + rot * sb[s * HD + d];
}

// k: rmsnorm+RoPE -> cache[kvh][sp+s]; v: copy -> cache
__global__ __launch_bounds__(128)
void kvprep_k(const float* __restrict__ kb, const float* __restrict__ vb,
              const float* __restrict__ w, const float* __restrict__ cb,
              const float* __restrict__ sb, const int* __restrict__ sp,
              float* __restrict__ kc, float* __restrict__ vc) {
  const int n = blockIdx.x, s = blockIdx.y, d = threadIdx.x;
  __shared__ float red[4];
  __shared__ float xr[128];
  const float x = kb[(size_t)s * 256 + n * HD + d];
  float sq = x * x;
  for (int m = 1; m < 32; m <<= 1) sq += __shfl_xor(sq, m);
  if ((threadIdx.x & 31) == 0) red[threadIdx.x >> 5] = sq;
  __syncthreads();
  const float inv = rsqrtf((red[0] + red[1] + red[2] + red[3]) / 128.f + 1e-6f);
  const float xn = x * inv * w[d];
  xr[d] = xn;
  __syncthreads();
  const float rot = (d < 64) ? -xr[d + 64] : xr[d - 64];
  const int pos = sp[0] + s;
  kc[((size_t)n * MAXKV + pos) * HD + d] =
      xn * cb[s * HD + d] + rot * sb[s * HD + d];
  vc[((size_t)n * MAXKV + pos) * HD + d] = vb[(size_t)s * 256 + n * HD + d];
}

// ---------------------------------------------------------------------------
// Flash attention, WMMA QK^T and PV, online softmax, fused sigmoid gate.
// grid (q-tile 16, head 16); 256 threads.
// ---------------------------------------------------------------------------
__global__ __launch_bounds__(256)
void attn_k(const float* __restrict__ Qb, const float* __restrict__ KC,
            const float* __restrict__ VC, const float* __restrict__ qg,
            const int* __restrict__ spp, float* __restrict__ ao) {
  const int qt = blockIdx.x, h = blockIdx.y;
  const int kvh = h >> 3;
  const int t = threadIdx.x, lane = t & 31, wv = t >> 5;
  const int l16 = lane & 15, hf = lane >> 4;
  const int base = spp[0];

  __shared__ __align__(16) bh Qs[64][136];
  __shared__ __align__(16) bh Ks[64][136];
  __shared__ __align__(16) bh Vt[128][72];   // transposed V: [d][j]
  __shared__ __align__(16) bh Ps[64][72];
  __shared__ float rmax[64], rsum[64], ralpha[64];
  __shared__ float pmax[2][64], psum[2][64];

  for (int i = t; i < 64 * 32; i += 256) {
    const int r = i >> 5, q = i & 31;
    const float4 v = *(const float4*)(Qb + ((size_t)h * S_LEN + qt * 64 + r) * HD + q * 4);
    bh* d = &Qs[r][q * 4];
    d[0] = (bh)v.x; d[1] = (bh)v.y; d[2] = (bh)v.z; d[3] = (bh)v.w;
  }
  if (t < 64) { rmax[t] = -1e30f; rsum[t] = 0.f; }
  __syncthreads();

  const int mgp = wv & 3, ncp = wv >> 2;  // PV wave mapping: 4 rowgroups x 2 col halves
  f8 oacc[4];
#pragma unroll
  for (int i = 0; i < 4; ++i)
#pragma unroll
    for (int e = 0; e < 8; ++e) oacc[i][e] = 0.f;

  const int ntiles = (base + qt * 64 + 64 + 63) >> 6;
  const float scale = 0.08838834764831845f;

  for (int kt = 0; kt < ntiles; ++kt) {
    for (int i = t; i < 64 * 32; i += 256) {
      const int r = i >> 5, q = i & 31;
      const float4 kv = *(const float4*)(KC + ((size_t)kvh * MAXKV + kt * 64 + r) * HD + q * 4);
      bh* d = &Ks[r][q * 4];
      d[0] = (bh)kv.x; d[1] = (bh)kv.y; d[2] = (bh)kv.z; d[3] = (bh)kv.w;
      const float4 vv = *(const float4*)(VC + ((size_t)kvh * MAXKV + kt * 64 + r) * HD + q * 4);
      Vt[q * 4 + 0][r] = (bh)vv.x; Vt[q * 4 + 1][r] = (bh)vv.y;
      Vt[q * 4 + 2][r] = (bh)vv.z; Vt[q * 4 + 3][r] = (bh)vv.w;
    }
    __syncthreads();

    const int mg = wv >> 1, ng2 = wv & 1;  // score mapping: 4 rowgroups x 2 col halves
    f8 sc[2];
#pragma unroll
    for (int nf = 0; nf < 2; ++nf)
#pragma unroll
      for (int e = 0; e < 8; ++e) sc[nf][e] = 0.f;
#pragma unroll
    for (int kk = 0; kk < 4; ++kk) {
      const int r = mg * 16 + l16;
      const bh16 aq = mk16(*(const bh8*)&Qs[r][kk * 32 + hf * 8],
                           *(const bh8*)&Qs[r][kk * 32 + 16 + hf * 8]);
#pragma unroll
      for (int nf = 0; nf < 2; ++nf) {
        const int cn = ng2 * 32 + nf * 16 + l16;
        const bh16 bk = mk16(*(const bh8*)&Ks[cn][kk * 32 + hf * 8],
                             *(const bh8*)&Ks[cn][kk * 32 + 16 + hf * 8]);
        sc[nf] = __builtin_amdgcn_wmma_f32_16x16x32_bf16(false, aq, false, bk,
                                                         (short)0, sc[nf], false, false);
      }
    }
    // scale + causal mask
#pragma unroll
    for (int nf = 0; nf < 2; ++nf) {
      const int tk = kt * 64 + ng2 * 32 + nf * 16 + l16;
#pragma unroll
      for (int r = 0; r < 8; ++r) {
        const int row = mg * 16 + r + 8 * hf;
        float v = sc[nf][r] * scale;
        if (tk > base + qt * 64 + row) v = -1e30f;
        sc[nf][r] = v;
      }
    }
#pragma unroll
    for (int r = 0; r < 8; ++r) {
      float m = fmaxf(sc[0][r], sc[1][r]);
      m = fmaxf(m, __shfl_xor(m, 1)); m = fmaxf(m, __shfl_xor(m, 2));
      m = fmaxf(m, __shfl_xor(m, 4)); m = fmaxf(m, __shfl_xor(m, 8));
      if (l16 == 0) pmax[ng2][mg * 16 + r + 8 * hf] = m;
    }
    __syncthreads();
    if (t < 64) {
      const float mo = rmax[t];
      const float mn = fmaxf(mo, fmaxf(pmax[0][t], pmax[1][t]));
      ralpha[t] = (mo > -1e29f) ? __expf(mo - mn) : 0.f;
      rmax[t] = mn;
    }
    __syncthreads();
#pragma unroll
    for (int nf = 0; nf < 2; ++nf) {
      const int col = ng2 * 32 + nf * 16 + l16;
#pragma unroll
      for (int r = 0; r < 8; ++r) {
        const int row = mg * 16 + r + 8 * hf;
        const float p = __expf(sc[nf][r] - rmax[row]);
        sc[nf][r] = p;
        Ps[row][col] = (bh)p;
      }
    }
#pragma unroll
    for (int r = 0; r < 8; ++r) {
      float s2 = sc[0][r] + sc[1][r];
      s2 += __shfl_xor(s2, 1); s2 += __shfl_xor(s2, 2);
      s2 += __shfl_xor(s2, 4); s2 += __shfl_xor(s2, 8);
      if (l16 == 0) psum[ng2][mg * 16 + r + 8 * hf] = s2;
    }
    __syncthreads();
    if (t < 64) rsum[t] = rsum[t] * ralpha[t] + psum[0][t] + psum[1][t];
#pragma unroll
    for (int nf = 0; nf < 4; ++nf)
#pragma unroll
      for (int r = 0; r < 8; ++r)
        oacc[nf][r] = oacc[nf][r] * ralpha[mgp * 16 + r + 8 * hf];
    // PV
#pragma unroll
    for (int kk = 0; kk < 2; ++kk) {
      const int r = mgp * 16 + l16;
      const bh16 ap = mk16(*(const bh8*)&Ps[r][kk * 32 + hf * 8],
                           *(const bh8*)&Ps[r][kk * 32 + 16 + hf * 8]);
#pragma unroll
      for (int nf = 0; nf < 4; ++nf) {
        const int cd = ncp * 64 + nf * 16 + l16;
        const bh16 bv = mk16(*(const bh8*)&Vt[cd][kk * 32 + hf * 8],
                             *(const bh8*)&Vt[cd][kk * 32 + 16 + hf * 8]);
        oacc[nf] = __builtin_amdgcn_wmma_f32_16x16x32_bf16(false, ap, false, bv,
                                                           (short)0, oacc[nf], false, false);
      }
    }
    __syncthreads();
  }
#pragma unroll
  for (int nf = 0; nf < 4; ++nf) {
    const int d = ncp * 64 + nf * 16 + l16;
#pragma unroll
    for (int r = 0; r < 8; ++r) {
      const int row = mgp * 16 + r + 8 * hf;
      const int s = qt * 64 + row;
      float v = oacc[nf][r] / rsum[row];
      v *= sigf(qg[(size_t)s * 4096 + h * 256 + 128 + d]);
      ao[(size_t)s * HID + h * HD + d] = v;
    }
  }
}

// ---------------------------------------------------------------------------
extern "C" void kernel_launch(void* const* d_in, const int* in_sizes, int n_in,
                              void* d_out, int out_size, void* d_ws, size_t ws_size,
                              hipStream_t stream) {
  (void)in_sizes; (void)n_in; (void)out_size; (void)ws_size;
  const int* ids = (const int*)d_in[0];
  const int* sp = (const int*)d_in[2];
  const float* rcos = (const float*)d_in[3];
  const float* rsin = (const float*)d_in[4];
  const float* cst = (const float*)d_in[6];
  const float* rst = (const float*)d_in[7];
  const float* kcache = (const float*)d_in[8];
  const float* vcache = (const float*)d_in[9];
  const float* embw = (const float*)d_in[10];
  const float* lnw1 = (const float*)d_in[11];
  const float* qkvw = (const float*)d_in[12];
  const float* convw = (const float*)d_in[13];
  const float* convb = (const float*)d_in[14];
  const float* zw = (const float*)d_in[15];
  const float* bw = (const float*)d_in[16];
  const float* aw = (const float*)d_in[17];
  const float* alog = (const float*)d_in[18];
  const float* dtb = (const float*)d_in[19];
  const float* dnw = (const float*)d_in[20];
  const float* outw = (const float*)d_in[21];
  const float* plnw1 = (const float*)d_in[22];
  const float* gw1 = (const float*)d_in[23];
  const float* uw1 = (const float*)d_in[24];
  const float* dw1 = (const float*)d_in[25];
  const float* lnw2 = (const float*)d_in[26];
  const float* qw = (const float*)d_in[27];
  const float* qnw = (const float*)d_in[28];
  const float* kw = (const float*)d_in[29];
  const float* knw = (const float*)d_in[30];
  const float* vw = (const float*)d_in[31];
  const float* ow = (const float*)d_in[32];
  const float* plnw2 = (const float*)d_in[33];
  const float* gw2 = (const float*)d_in[34];
  const float* uw2 = (const float*)d_in[35];
  const float* dw2 = (const float*)d_in[36];
  const float* fnw = (const float*)d_in[37];
  const float* lmw = (const float*)d_in[38];

  float* logits = (float*)d_out;
  float* o_conv = logits + (size_t)S_LEN * 32000;
  float* o_rec = o_conv + (size_t)CDIM * 4;
  float* o_kc = o_rec + (size_t)NVH * 128 * 128;
  float* o_vc = o_kc + (size_t)NKVH * MAXKV * HD;

  // workspace layout (floats)
  const size_t SH = (size_t)S_LEN * HID;   // 2,097,152
  const size_t SC = (size_t)S_LEN * CDIM;  // 4,194,304
  const size_t SF = (size_t)S_LEN * DFF;   // 6,291,456
  float* wp = (float*)d_ws;
  size_t off = 0;
  float* hidden = wp + off; off += SH;
  float* xb = wp + off; off += SH;
  float* mpre = wp + off; off += SC;
  float* mixed = wp + off; off += SC;
  float* zb = wp + off; off += SH;
  float* corex = wp + off; off += SH;
  float* coren = wp + off; off += SH;
  float* gt = wp + off; off += SF;
  float* ut = wp + off; off += SF;
  float* qg = wp + off; off += SC;
  float* Qb = wp + off; off += SH;  // 16*1024*128
  float* kbuf = wp + off; off += (size_t)S_LEN * 256;
  float* vbuf = wp + off; off += (size_t)S_LEN * 256;
  float* ao = wp + off; off += SH;
  float* betab = wp + off; off += (size_t)S_LEN * 16;
  float* gb = wp + off; off += (size_t)S_LEN * 16;

  // ---- delta-rule (linear attention) layer ----
  embed_k<<<S_LEN, 256, 0, stream>>>(ids, embw, hidden);
  rmsnorm_k<<<S_LEN, 256, 0, stream>>>(hidden, lnw1, xb, HID);
  gemm_k<0><<<dim3(CDIM / 128, 8), 256, 0, stream>>>(xb, qkvw, mpre, nullptr, S_LEN, CDIM, HID);
  conv_k<<<dim3(16, S_LEN), 256, 0, stream>>>(mpre, cst, convw, convb, mixed);
  newconv_k<<<16, 256, 0, stream>>>(mpre, o_conv);
  gemm_k<0><<<dim3(HID / 128, 8), 256, 0, stream>>>(xb, zw, zb, nullptr, S_LEN, HID, HID);
  ba_k<<<S_LEN, 256, 0, stream>>>(xb, bw, aw, alog, dtb, betab, gb);
  delta_k<<<64, 256, 0, stream>>>(mixed, gb, betab, rst, corex, o_rec);
  gatednorm_k<<<dim3(NVH, S_LEN), 128, 0, stream>>>(corex, zb, dnw, coren);
  gemm_k<1><<<dim3(HID / 128, 8), 256, 0, stream>>>(coren, outw, hidden, nullptr, S_LEN, HID, HID);
  rmsnorm_k<<<S_LEN, 256, 0, stream>>>(hidden, plnw1, xb, HID);
  gemm_k<2><<<dim3(DFF / 128, 8), 256, 0, stream>>>(xb, gw1, gt, nullptr, S_LEN, DFF, HID);
  gemm_k<3><<<dim3(DFF / 128, 8), 256, 0, stream>>>(xb, uw1, ut, gt, S_LEN, DFF, HID);
  gemm_k<1><<<dim3(HID / 128, 8), 256, 0, stream>>>(ut, dw1, hidden, nullptr, S_LEN, HID, DFF);

  // ---- full attention layer ----
  rmsnorm_k<<<S_LEN, 256, 0, stream>>>(hidden, lnw2, xb, HID);
  gemm_k<0><<<dim3(4096 / 128, 8), 256, 0, stream>>>(xb, qw, qg, nullptr, S_LEN, 4096, HID);
  gemm_k<0><<<dim3(2, 8), 256, 0, stream>>>(xb, kw, kbuf, nullptr, S_LEN, 256, HID);
  gemm_k<0><<<dim3(2, 8), 256, 0, stream>>>(xb, vw, vbuf, nullptr, S_LEN, 256, HID);
  qprep_k<<<dim3(NQH, S_LEN), 128, 0, stream>>>(qg, qnw, rcos, rsin, Qb);
  (void)hipMemcpyAsync(o_kc, kcache, (size_t)NKVH * MAXKV * HD * sizeof(float),
                       hipMemcpyDeviceToDevice, stream);
  (void)hipMemcpyAsync(o_vc, vcache, (size_t)NKVH * MAXKV * HD * sizeof(float),
                       hipMemcpyDeviceToDevice, stream);
  kvprep_k<<<dim3(NKVH, S_LEN), 128, 0, stream>>>(kbuf, vbuf, knw, rcos, rsin, sp, o_kc, o_vc);
  attn_k<<<dim3(S_LEN / 64, NQH), 256, 0, stream>>>(Qb, o_kc, o_vc, qg, sp, ao);
  gemm_k<1><<<dim3(HID / 128, 8), 256, 0, stream>>>(ao, ow, hidden, nullptr, S_LEN, HID, HID);
  rmsnorm_k<<<S_LEN, 256, 0, stream>>>(hidden, plnw2, xb, HID);
  gemm_k<2><<<dim3(DFF / 128, 8), 256, 0, stream>>>(xb, gw2, gt, nullptr, S_LEN, DFF, HID);
  gemm_k<3><<<dim3(DFF / 128, 8), 256, 0, stream>>>(xb, uw2, ut, gt, S_LEN, DFF, HID);
  gemm_k<1><<<dim3(HID / 128, 8), 256, 0, stream>>>(ut, dw2, hidden, nullptr, S_LEN, HID, DFF);

  // ---- LM head ----
  rmsnorm_k<<<S_LEN, 256, 0, stream>>>(hidden, fnw, xb, HID);
  gemm_k<0><<<dim3(32000 / 128, 8), 256, 0, stream>>>(xb, lmw, logits, nullptr, S_LEN, 32000, HID);
}
